// MoEDSV2_42322607735340
// MI455X (gfx1250) — compile-verified, matrix operand order checked by reading, not run on hardware
//
#include <hip/hip_runtime.h>

// ---------------- problem constants (from reference) ----------------
#define DD    2048      // model dim
#define EE    32        // experts
#define FF    1408      // expert inter dim
#define SS    2816      // shared expert inter dim (= 2*FF)
#define TT    4096      // tokens (B*N)
#define TOPK  6
#define CAP   1152      // int(1.5 * T * K / E)
#define MROWS 32        // rows per block (two 16-row WMMA subtiles)
#define LDH   (FF + 8)  // LDS stride for h tile (f16 elems)
#define LDA   (DD + 8)  // LDS stride for A tile (f16 elems)

typedef __attribute__((ext_vector_type(16))) _Float16 v16h;
typedef __attribute__((ext_vector_type(8)))  float    v8f;
typedef __attribute__((ext_vector_type(2)))  __fp16   h2v;   // matches cvt_pkrtz return type

union Frag {
    v16h v;
    h2v  h[8];
    uint4 q[2];
};

// D = A(16x32) * B(32x16) + C via CDNA5 WMMA, f16 in / f32 acc
__device__ __forceinline__ v8f wmma16(v16h a, v16h b, v8f c) {
    return __builtin_amdgcn_wmma_f32_16x16x32_f16(
        /*neg_a=*/false, a, /*neg_b=*/false, b,
        /*c_mod=*/(short)0, c, /*reuse_a=*/false, /*reuse_b=*/false);
}

// fast silu: g * v_rcp_f32(1 + exp(-g))  (avoids IEEE v_div_scale sequence)
__device__ __forceinline__ float fast_silu(float g) {
    return g * __builtin_amdgcn_rcpf(1.f + __expf(-g));
}

// A/B fragment from a row-major f16 matrix tile (LDS or global).
// lane L: row = L%16, k-chunks at kbase=(L/16)*8 and kbase+16 (ISA 7.12.2).
__device__ __forceinline__ v16h frag_f16(const _Float16* __restrict__ base,
                                         int ldk, int lane, int k0) {
    int row = lane & 15;
    int kb  = ((lane >> 4) << 3) + k0;
    const _Float16* p = base + (size_t)row * ldk + kb;
    Frag f;
    f.q[0] = *(const uint4*)(p);
    f.q[1] = *(const uint4*)(p + 16);
    return f.v;
}

// same fragment from a row-major f32 matrix, converted to f16 in registers
// with v_cvt_pk_rtz_f16_f32 (avoids a separate weight-conversion pass).
__device__ __forceinline__ v16h frag_f32(const float* __restrict__ base,
                                         size_t ldk, int lane, int k0) {
    int row = lane & 15;
    int kb  = ((lane >> 4) << 3) + k0;
    const float* p = base + (size_t)row * ldk + kb;
    float4 a0 = *(const float4*)(p);
    float4 a1 = *(const float4*)(p + 4);
    float4 b0 = *(const float4*)(p + 16);
    float4 b1 = *(const float4*)(p + 20);
    Frag f;
    f.h[0] = __builtin_amdgcn_cvt_pkrtz(a0.x, a0.y);
    f.h[1] = __builtin_amdgcn_cvt_pkrtz(a0.z, a0.w);
    f.h[2] = __builtin_amdgcn_cvt_pkrtz(a1.x, a1.y);
    f.h[3] = __builtin_amdgcn_cvt_pkrtz(a1.z, a1.w);
    f.h[4] = __builtin_amdgcn_cvt_pkrtz(b0.x, b0.y);
    f.h[5] = __builtin_amdgcn_cvt_pkrtz(b0.z, b0.w);
    f.h[6] = __builtin_amdgcn_cvt_pkrtz(b1.x, b1.y);
    f.h[7] = __builtin_amdgcn_cvt_pkrtz(b1.z, b1.w);
    return f.v;
}

// async copy of one 16B chunk global -> LDS (ASYNCcnt tracked; CDNA5 §15.18.3)
__device__ __forceinline__ void async_copy_b128(unsigned lds_byte_addr,
                                                const _Float16* gsrc) {
    asm volatile("global_load_async_to_lds_b128 %0, %1, off"
                 :: "v"(lds_byte_addr), "v"(gsrc) : "memory");
}
__device__ __forceinline__ void wait_async() {
    asm volatile("s_wait_asynccnt 0x0" ::: "memory");
}

// ---------------- kernel 1: reset routing metadata + aux output ----------------
__global__ __launch_bounds__(256) void moe_init_meta(int* __restrict__ counts,
                                                     int* __restrict__ slot_token,
                                                     float* __restrict__ aux_out) {
    int i = blockIdx.x * 256 + threadIdx.x;
    if (i < EE) counts[i] = 0;
    if (i < EE * CAP) slot_token[i] = -1;
    if (i == 0) *aux_out = 0.0f;
}

// ---------------- kernel 2: zero y (atomic accumulation target) ----------------
__global__ __launch_bounds__(256) void moe_zero_y(float* __restrict__ y) {
    size_t i = ((size_t)blockIdx.x * 256 + threadIdx.x) * 4;
    *(float4*)(y + i) = make_float4(0.f, 0.f, 0.f, 0.f);
}

// ---------------- kernel 3: x f32 -> f16 staging ----------------
__global__ __launch_bounds__(256) void moe_convert_x(const float* __restrict__ x,
                                                     _Float16* __restrict__ xb) {
    size_t i = ((size_t)blockIdx.x * 256 + threadIdx.x) * 8;
    float4 a = *(const float4*)(x + i);
    float4 b = *(const float4*)(x + i + 4);
    Frag f;
    f.h[0] = __builtin_amdgcn_cvt_pkrtz(a.x, a.y);
    f.h[1] = __builtin_amdgcn_cvt_pkrtz(a.z, a.w);
    f.h[2] = __builtin_amdgcn_cvt_pkrtz(b.x, b.y);
    f.h[3] = __builtin_amdgcn_cvt_pkrtz(b.z, b.w);
    *(uint4*)(xb + i) = f.q[0];
}

// ---------------- kernel 4: gate (softmax + top-6 + dispatch) ----------------
__global__ __launch_bounds__(256) void moe_gate(const float* __restrict__ x,
                                                const float* __restrict__ wg,
                                                int* __restrict__ counts,
                                                int* __restrict__ slot_token,
                                                float* __restrict__ slot_weight) {
    int lane  = threadIdx.x & 31;
    int wv    = threadIdx.x >> 5;
    int token = blockIdx.x * 8 + wv;

    const float* xr = x  + (size_t)token * DD;
    const float* wr = wg + (size_t)lane  * DD;
    float dot = 0.f;
    for (int d = 0; d < DD; d += 4) {
        float4 xv  = *(const float4*)(xr + d);
        float4 wv4 = *(const float4*)(wr + d);
        dot += xv.x * wv4.x + xv.y * wv4.y + xv.z * wv4.z + xv.w * wv4.w;
    }
    float mx = dot;
    for (int off = 16; off > 0; off >>= 1) mx = fmaxf(mx, __shfl_xor(mx, off, 32));
    float ex = __expf(dot - mx);
    float sm = ex;
    for (int off = 16; off > 0; off >>= 1) sm += __shfl_xor(sm, off, 32);
    float prob = ex / sm;

    float p = prob;
    for (int k = 0; k < TOPK; k++) {
        float v = p;
        int who = lane;
        for (int off = 16; off > 0; off >>= 1) {
            float ov = __shfl_xor(v,   off, 32);
            int   ow = __shfl_xor(who, off, 32);
            if (ov > v || (ov == v && ow < who)) { v = ov; who = ow; }
        }
        float wsel = __shfl(prob, who, 32);
        if (lane == 0) {
            int pos = atomicAdd(&counts[who], 1);
            if (pos < CAP) {
                slot_token[who * CAP + pos]  = token;
                slot_weight[who * CAP + pos] = wsel;
            }
        }
        if (lane == who) p = -3.0e38f;
    }
}

// ---------------- kernel 5: routed expert fused SwiGLU FFN (M=32) ----------------
__global__ __launch_bounds__(256) void moe_routed_ffn(
    const _Float16* __restrict__ xb,
    const float* __restrict__ w1, const float* __restrict__ w2,
    const float* __restrict__ w3,
    const int* __restrict__ counts, const int* __restrict__ slot_token,
    const float* __restrict__ slot_weight, float* __restrict__ y) {

    const int MT = CAP / MROWS;          // 36 m-tiles per expert
    int e  = blockIdx.x / MT;
    int mt = blockIdx.x % MT;
    int cnt = counts[e];
    if (cnt > CAP) cnt = CAP;
    if (mt * MROWS >= cnt) return;

    __shared__ __align__(16) _Float16 Ax[MROWS * LDA];   // 131.5 KB A tile
    __shared__ __align__(16) _Float16 Hs[MROWS * LDH];   //  90.6 KB h tile
    __shared__ int   stok[MROWS];
    __shared__ float swt[MROWS];

    int tid = threadIdx.x;
    if (tid < MROWS) {
        int s = mt * MROWS + tid;
        int t = slot_token[e * CAP + s];
        stok[tid] = t;
        swt[tid]  = (t >= 0) ? slot_weight[e * CAP + s] : 0.f;
    }
    __syncthreads();

    // ---- stage A tile (32 gathered token rows) into LDS via async copies ----
    {
        int r = tid >> 3, c = tid & 7;               // 8 threads per row
        int t = stok[r];
        _Float16* dst = &Ax[r * LDA + c * 256];
        if (t >= 0) {
            const _Float16* src = xb + (size_t)t * DD + c * 256;
            #pragma unroll
            for (int j = 0; j < 32; j++)
                async_copy_b128((unsigned)(size_t)(dst + j * 8), src + j * 8);
        } else {
            uint4 z = make_uint4(0, 0, 0, 0);
            #pragma unroll
            for (int j = 0; j < 32; j++)
                *(uint4*)(dst + j * 8) = z;
        }
        wait_async();
    }
    __syncthreads();

    int lane = tid & 31, wave = tid >> 5;
    const float* w1e = w1 + (size_t)e * FF * DD;
    const float* w3e = w3 + (size_t)e * FF * DD;
    const float* w2e = w2 + (size_t)e * DD * FF;

    // ---- phase 1: h = silu(x w1^T) * (x w3^T) over F ----
    for (int nt = wave; nt < FF / 16; nt += 8) {
        if (nt + 8 < FF / 16) {
            __builtin_prefetch(w1e + (size_t)((nt + 8) * 16) * DD, 0, 1);
            __builtin_prefetch(w3e + (size_t)((nt + 8) * 16) * DD, 0, 1);
        }
        v8f a1lo = {}, a1hi = {}, a3lo = {}, a3hi = {};
        const float* b1 = w1e + (size_t)(nt * 16) * DD;
        const float* b3 = w3e + (size_t)(nt * 16) * DD;
        for (int k0 = 0; k0 < DD; k0 += 32) {
            v16h alo = frag_f16(Ax,            LDA, lane, k0);
            v16h ahi = frag_f16(Ax + 16 * LDA, LDA, lane, k0);
            v16h f1  = frag_f32(b1, DD, lane, k0);
            v16h f3  = frag_f32(b3, DD, lane, k0);
            a1lo = wmma16(alo, f1, a1lo);
            a1hi = wmma16(ahi, f1, a1hi);     // B reused across row groups
            a3lo = wmma16(alo, f3, a3lo);
            a3hi = wmma16(ahi, f3, a3hi);
        }
        int n  = nt * 16 + (lane & 15);
        int mh = (lane >> 4) << 3;
        #pragma unroll
        for (int i = 0; i < 8; i++) {
            Hs[(size_t)(i + mh) * LDH + n]      = (_Float16)(fast_silu(a1lo[i]) * a3lo[i]);
            Hs[(size_t)(16 + i + mh) * LDH + n] = (_Float16)(fast_silu(a1hi[i]) * a3hi[i]);
        }
    }
    __syncthreads();

    // ---- phase 2: out = h @ w2^T, weighted atomic combine ----
    for (int nt = wave; nt < DD / 16; nt += 8) {
        if (nt + 8 < DD / 16)
            __builtin_prefetch(w2e + (size_t)((nt + 8) * 16) * FF, 0, 1);
        v8f acclo = {}, acchi = {};
        const float* bb = w2e + (size_t)(nt * 16) * FF;
        for (int k0 = 0; k0 < FF; k0 += 32) {
            v16h b = frag_f32(bb, FF, lane, k0);
            acclo = wmma16(frag_f16(Hs,            LDH, lane, k0), b, acclo);
            acchi = wmma16(frag_f16(Hs + 16 * LDH, LDH, lane, k0), b, acchi);
        }
        int n  = nt * 16 + (lane & 15);
        int mh = (lane >> 4) << 3;
        #pragma unroll
        for (int i = 0; i < 8; i++) {
            int m0 = i + mh;
            int m1 = 16 + i + mh;
            int t0 = stok[m0];
            int t1 = stok[m1];
            if (t0 >= 0) atomicAdd(&y[(size_t)t0 * DD + n], acclo[i] * swt[m0]);
            if (t1 >= 0) atomicAdd(&y[(size_t)t1 * DD + n], acchi[i] * swt[m1]);
        }
    }
}

// ---------------- kernel 6: shared expert fused SwiGLU FFN (M=32) ----------------
__global__ __launch_bounds__(256) void moe_shared_ffn(
    const _Float16* __restrict__ xb,
    const float* __restrict__ ws1, const float* __restrict__ ws2,
    const float* __restrict__ ws3, float* __restrict__ y) {

    int mt = blockIdx.x >> 1;           // 128 m-tiles of 32 tokens
    int c  = blockIdx.x & 1;            // chunk of S

    __shared__ __align__(16) _Float16 Ax[MROWS * LDA];
    __shared__ __align__(16) _Float16 Hs[MROWS * LDH];

    int tid  = threadIdx.x;

    // ---- stage A tile into LDS via async copies ----
    {
        int r = tid >> 3, cc = tid & 7;
        _Float16* dst = &Ax[r * LDA + cc * 256];
        const _Float16* src = xb + (size_t)(mt * MROWS + r) * DD + cc * 256;
        #pragma unroll
        for (int j = 0; j < 32; j++)
            async_copy_b128((unsigned)(size_t)(dst + j * 8), src + j * 8);
        wait_async();
    }
    __syncthreads();

    int lane = tid & 31, wave = tid >> 5;
    const float* w1c = ws1 + (size_t)(c * FF) * DD;
    const float* w3c = ws3 + (size_t)(c * FF) * DD;

    // ---- phase 1 ----
    for (int nt = wave; nt < FF / 16; nt += 8) {
        if (nt + 8 < FF / 16) {
            __builtin_prefetch(w1c + (size_t)((nt + 8) * 16) * DD, 0, 1);
            __builtin_prefetch(w3c + (size_t)((nt + 8) * 16) * DD, 0, 1);
        }
        v8f a1lo = {}, a1hi = {}, a3lo = {}, a3hi = {};
        const float* b1 = w1c + (size_t)(nt * 16) * DD;
        const float* b3 = w3c + (size_t)(nt * 16) * DD;
        for (int k0 = 0; k0 < DD; k0 += 32) {
            v16h alo = frag_f16(Ax,            LDA, lane, k0);
            v16h ahi = frag_f16(Ax + 16 * LDA, LDA, lane, k0);
            v16h f1  = frag_f32(b1, DD, lane, k0);
            v16h f3  = frag_f32(b3, DD, lane, k0);
            a1lo = wmma16(alo, f1, a1lo);
            a1hi = wmma16(ahi, f1, a1hi);
            a3lo = wmma16(alo, f3, a3lo);
            a3hi = wmma16(ahi, f3, a3hi);
        }
        int n  = nt * 16 + (lane & 15);
        int mh = (lane >> 4) << 3;
        #pragma unroll
        for (int i = 0; i < 8; i++) {
            Hs[(size_t)(i + mh) * LDH + n]      = (_Float16)(fast_silu(a1lo[i]) * a3lo[i]);
            Hs[(size_t)(16 + i + mh) * LDH + n] = (_Float16)(fast_silu(a1hi[i]) * a3hi[i]);
        }
    }
    __syncthreads();

    // ---- phase 2: y += h_chunk @ ws2[:, chunk]^T ----
    for (int nt = wave; nt < DD / 16; nt += 8) {
        v8f acclo = {}, acchi = {};
        const float* bb = ws2 + (size_t)(nt * 16) * SS + (size_t)c * FF;
        for (int k0 = 0; k0 < FF; k0 += 32) {
            v16h b = frag_f32(bb, SS, lane, k0);
            acclo = wmma16(frag_f16(Hs,            LDH, lane, k0), b, acclo);
            acchi = wmma16(frag_f16(Hs + 16 * LDH, LDH, lane, k0), b, acchi);
        }
        int n  = nt * 16 + (lane & 15);
        int mh = (lane >> 4) << 3;
        #pragma unroll
        for (int i = 0; i < 8; i++) {
            int t0 = mt * MROWS + i + mh;
            int t1 = mt * MROWS + 16 + i + mh;
            atomicAdd(&y[(size_t)t0 * DD + n], acclo[i]);
            atomicAdd(&y[(size_t)t1 * DD + n], acchi[i]);
        }
    }
}

// ---------------- host launcher ----------------
extern "C" void kernel_launch(void* const* d_in, const int* in_sizes, int n_in,
                              void* d_out, int out_size, void* d_ws, size_t ws_size,
                              hipStream_t stream) {
    (void)in_sizes; (void)n_in; (void)ws_size;
    const float* x   = (const float*)d_in[0];
    const float* wg  = (const float*)d_in[1];
    const float* w1  = (const float*)d_in[2];
    const float* w2  = (const float*)d_in[3];
    const float* w3  = (const float*)d_in[4];
    const float* ws1 = (const float*)d_in[5];
    const float* ws2 = (const float*)d_in[6];
    const float* ws3 = (const float*)d_in[7];
    float* y = (float*)d_out;

    // workspace layout: xb f16 [T*D] | counts [E] (padded) | slot_token | slot_weight
    char* ws = (char*)d_ws;
    _Float16* xb   = (_Float16*)ws;
    int* counts    = (int*)(ws + (size_t)TT * DD * 2);
    int* slot_tok  = counts + 64;
    float* slot_wt = (float*)(slot_tok + EE * CAP);

    moe_init_meta<<<(EE * CAP + 255) / 256, 256, 0, stream>>>(counts, slot_tok,
                                                              y + (out_size - 1));
    moe_zero_y<<<(TT * DD) / (256 * 4), 256, 0, stream>>>(y);
    moe_convert_x<<<(TT * DD) / (256 * 8), 256, 0, stream>>>(x, xb);
    moe_gate<<<TT / 8, 256, 0, stream>>>(x, wg, counts, slot_tok, slot_wt);
    moe_shared_ffn<<<(TT / MROWS) * 2, 256, 0, stream>>>(xb, ws1, ws2, ws3, y);
    moe_routed_ffn<<<EE * (CAP / MROWS), 256, 0, stream>>>(xb, w1, w2, w3,
                                                           counts, slot_tok, slot_wt, y);
}
